// MCLMask_80461917323852
// MI455X (gfx1250) — matrix operand order repaired and checked by reading, not compiled, not run on hardware
//
#include <hip/hip_runtime.h>
#include <hip/hip_bf16.h>
#include <math.h>

// ---------------- problem constants (from reference) ----------------
#define BB    4      // batch
#define QQ    75     // queries per batch
#define NN    300    // BB*QQ independent problems
#define CH    640    // channels (K of big GEMM)
#define MM    100    // spatial positions (h*w)
#define NWAY  5
#define KSHOT 5
#define MS    500    // n_way * MM (support positions)
#define MSP   512    // MS padded to multiple of 16
#define MQP   128    // MM padded to multiple of 32 (2-tile M blocking)

typedef __attribute__((ext_vector_type(16))) _Float16 v16h;
typedef __attribute__((ext_vector_type(8)))  _Float16 h8;
typedef __attribute__((ext_vector_type(8)))  float    v8f;

// ---------------- K1: support mean over k_shot, L2-normalize, f16 B^T layout ----------------
// sup_h layout: [b][row=n*100+s (512 rows, 500..511 zero)][c=640] contiguous in c
__global__ void k_prep_sup(const float* __restrict__ sup, float* __restrict__ supm,
                           _Float16* __restrict__ sup_h) {
  int b = blockIdx.x / NWAY, n = blockIdx.x % NWAY;
  int tid = threadIdx.x;
  if (tid < MM) {
    int m = tid;
    float ss = 0.f;
    for (int c = 0; c < CH; ++c) {
      float acc = 0.f;
      for (int k = 0; k < KSHOT; ++k)
        acc += sup[(((size_t)(b * 25 + n * KSHOT + k)) * CH + c) * MM + m];
      acc *= (1.f / KSHOT);
      supm[(((size_t)(b * NWAY + n)) * CH + c) * MM + m] = acc;
      ss += acc * acc;
    }
    float inv = 1.f / (1e-16f + sqrtf(ss));
    size_t row = (size_t)b * MSP + n * MM + m;
    for (int c = 0; c < CH; ++c)
      sup_h[row * CH + c] =
          (_Float16)(supm[(((size_t)(b * NWAY + n)) * CH + c) * MM + m] * inv);
  } else if (n == 0 && tid < MM + (MSP - MS)) {   // zero pad rows 500..511 (once per b)
    size_t row = (size_t)b * MSP + MS + (tid - MM);
    for (int c = 0; c < CH; ++c) sup_h[row * CH + c] = (_Float16)0.f;
  }
}

// ---------------- K2: query L2-normalize, f16 A layout [n][row=m (128)][c=640] ----------------
__global__ void k_prep_qry(const float* __restrict__ qry, _Float16* __restrict__ qry_h) {
  int n = blockIdx.x;            // 0..299
  int tid = threadIdx.x;         // 128 threads = 128 rows
  size_t base = (size_t)n * CH * MM;  // qry flat: (n*640 + c)*100 + m
  if (tid < MM) {
    float ss = 0.f;
    for (int c = 0; c < CH; ++c) {
      float v = qry[base + (size_t)c * MM + tid];
      ss += v * v;
    }
    float inv = 1.f / (1e-16f + sqrtf(ss));
    size_t row = (size_t)n * MQP + tid;
    for (int c = 0; c < CH; ++c)
      qry_h[row * CH + c] = (_Float16)(qry[base + (size_t)c * MM + tid] * inv);
  } else {                       // zero pad rows 100..127
    size_t row = (size_t)n * MQP + tid;
    for (int c = 0; c < CH; ++c) qry_h[row * CH + c] = (_Float16)0.f;
  }
}

// ---------------- WMMA fragment helpers (contiguous-K order, same for A and B, wave32) ----------------
// A 16x32 f16: lanes 0-15 = row M=lane, K 0..15 contiguous; lanes 16-31 = row M=lane-16, K 16..31
// B 32x16 f16: lanes 0-15 = col N=lane, K 0..15 contiguous; lanes 16-31 = col N=lane-16, K 16..31
__device__ __forceinline__ v16h load_fragA(const _Float16* rowp, int k, int lane) {
  int ka = (lane < 16) ? 0 : 16;
  h8 lo = *(const h8*)(rowp + k + ka);
  h8 hi = *(const h8*)(rowp + k + ka + 8);
  v16h r;
#pragma unroll
  for (int i = 0; i < 8; ++i) { r[i] = lo[i]; r[8 + i] = hi[i]; }
  return r;
}
__device__ __forceinline__ v16h load_fragB(const _Float16* rowp, int k, int lane) {
  int kb = (lane < 16) ? 0 : 16;
  h8 lo = *(const h8*)(rowp + k + kb);
  h8 hi = *(const h8*)(rowp + k + kb + 8);
  v16h r;
#pragma unroll
  for (int i = 0; i < 8; ++i) { r[i] = lo[i]; r[8 + i] = hi[i]; }
  return r;
}

// ---------------- K3: S = qry_n^T @ sup_n  (100x500, K=640) per n ----------------
// Register-blocked 2 M-tiles x 4 N-tiles per wave: 8 WMMAs per 12 b128 loads per K-step.
// S buffer: [n][128][512] f32
__global__ void k_gemm_S(const _Float16* __restrict__ qry_h,
                         const _Float16* __restrict__ sup_h,
                         float* __restrict__ S) {
  int wid = blockIdx.x * (blockDim.x >> 5) + (threadIdx.x >> 5);
  if (wid >= NN * 4 * 8) return;           // wave-uniform guard; EXEC stays all-1s
  int lane = threadIdx.x & 31;
  int n = wid >> 5, t = wid & 31;
  int bm = t >> 3, bn = t & 7;             // bm: 32-row block (0..3), bn: 64-col block (0..7)
  int b = n / QQ;
  const _Float16* Ap0 = qry_h + ((size_t)n * MQP + bm * 32 + (lane & 15)) * CH;
  const _Float16* Ap1 = Ap0 + (size_t)16 * CH;
  const _Float16* Bp0 = sup_h + ((size_t)b * MSP + bn * 64 + (lane & 15)) * CH;
  v8f acc[2][4];
#pragma unroll
  for (int i = 0; i < 2; ++i)
#pragma unroll
    for (int j = 0; j < 4; ++j) acc[i][j] = (v8f){};

  for (int k = 0; k < CH; k += 32) {
    v16h a0 = load_fragA(Ap0, k, lane);
    v16h a1 = load_fragA(Ap1, k, lane);
    v16h b0 = load_fragB(Bp0, k, lane);
    v16h b1 = load_fragB(Bp0 + (size_t)16 * CH, k, lane);
    v16h b2 = load_fragB(Bp0 + (size_t)32 * CH, k, lane);
    v16h b3 = load_fragB(Bp0 + (size_t)48 * CH, k, lane);
    acc[0][0] = __builtin_amdgcn_wmma_f32_16x16x32_f16(false, a0, false, b0, (short)0, acc[0][0], false, false);
    acc[0][1] = __builtin_amdgcn_wmma_f32_16x16x32_f16(false, a0, false, b1, (short)0, acc[0][1], false, false);
    acc[0][2] = __builtin_amdgcn_wmma_f32_16x16x32_f16(false, a0, false, b2, (short)0, acc[0][2], false, false);
    acc[0][3] = __builtin_amdgcn_wmma_f32_16x16x32_f16(false, a0, false, b3, (short)0, acc[0][3], false, false);
    acc[1][0] = __builtin_amdgcn_wmma_f32_16x16x32_f16(false, a1, false, b0, (short)0, acc[1][0], false, false);
    acc[1][1] = __builtin_amdgcn_wmma_f32_16x16x32_f16(false, a1, false, b1, (short)0, acc[1][1], false, false);
    acc[1][2] = __builtin_amdgcn_wmma_f32_16x16x32_f16(false, a1, false, b2, (short)0, acc[1][2], false, false);
    acc[1][3] = __builtin_amdgcn_wmma_f32_16x16x32_f16(false, a1, false, b3, (short)0, acc[1][3], false, false);
  }

  int r0  = bm * 32 + ((lane < 16) ? 0 : 8);
  int c0  = bn * 64 + (lane & 15);
#pragma unroll
  for (int i = 0; i < 2; ++i) {
#pragma unroll
    for (int j = 0; j < 4; ++j) {
      float* Sp = S + ((size_t)n * MQP + r0 + i * 16) * MSP + c0 + j * 16;
#pragma unroll
      for (int r = 0; r < 8; ++r) Sp[(size_t)r * MSP] = acc[i][j][r];
    }
  }
}

// ---------------- K4a: per-row online softmax stats of 20*S (max, sumexp) ----------------
__global__ void k_row_stats(const float* __restrict__ S,
                            float* __restrict__ rowm, float* __restrict__ rowz) {
  int wid = blockIdx.x * (blockDim.x >> 5) + (threadIdx.x >> 5);
  if (wid >= NN * MM) return;
  int lane = threadIdx.x & 31;
  int n = wid / MM, q = wid % MM;
  const float* Sp = S + ((size_t)n * MQP + q) * MSP;
  float m = -1e30f, z = 0.f;
  for (int s = lane; s < MS; s += 32) {
    float v = 20.f * Sp[s];
    float mn = fmaxf(m, v);
    z = z * __expf(m - mn) + __expf(v - mn);
    m = mn;
  }
  for (int off = 16; off; off >>= 1) {
    float mo = __shfl_xor(m, off, 32);
    float zo = __shfl_xor(z, off, 32);
    float mn = fmaxf(m, mo);
    z = z * __expf(m - mn) + zo * __expf(mo - mn);
    m = mn;
  }
  if (lane == 0) {
    rowm[(size_t)n * MQP + q] = m;
    rowz[(size_t)n * MQP + q] = z;
  }
}

// ---------------- K4b: per-col online softmax stats of 10*S ----------------
__global__ void k_col_stats(const float* __restrict__ S,
                            float* __restrict__ colm, float* __restrict__ colz) {
  int idx = blockIdx.x * blockDim.x + threadIdx.x;
  if (idx >= NN * MSP) return;
  int n = idx >> 9, s = idx & 511;
  float m = -1e30f, z = 0.f;
  if (s < MS) {
    const float* Sp = S + (size_t)n * MQP * MSP + s;
    for (int q = 0; q < MM; ++q) {
      float v = 10.f * Sp[(size_t)q * MSP];
      float mn = fmaxf(m, v);
      z = z * __expf(m - mn) + __expf(v - mn);
      m = mn;
    }
  }
  colm[(size_t)n * MSP + s] = m;
  colz[(size_t)n * MSP + s] = z;
}

// ---------------- K5: P[q,s]=T_qs[s,q], Q[q,s]=T_sq[q,s]  (f16, zero-padded) ----------------
__global__ void k_make_pq(const float* __restrict__ S,
                          const float* __restrict__ rowm, const float* __restrict__ rowz,
                          const float* __restrict__ colm, const float* __restrict__ colz,
                          _Float16* __restrict__ P, _Float16* __restrict__ Qm) {
  size_t idx = (size_t)blockIdx.x * blockDim.x + threadIdx.x;
  if (idx >= (size_t)NN * MQP * MSP) return;
  int s = (int)(idx & 511);
  size_t t = idx >> 9;
  int q = (int)(t & (MQP - 1));
  int n = (int)(t / MQP);
  float p = 0.f, qv = 0.f;
  if (q < MM && s < MS) {
    float sv = S[idx];
    p  = __expf(10.f * sv - colm[(size_t)n * MSP + s]) / colz[(size_t)n * MSP + s];
    qv = __expf(20.f * sv - rowm[(size_t)n * MQP + q]) / rowz[(size_t)n * MQP + q];
  }
  P[idx]  = (_Float16)p;
  Qm[idx] = (_Float16)qv;
}

// ---------------- K6: W = P @ Q^T  (100x100, K=512) per n; 7x7 WMMA tiles ----------------
__global__ void k_gemm_W(const _Float16* __restrict__ P, const _Float16* __restrict__ Qm,
                         float* __restrict__ W) {
  int wid = blockIdx.x * (blockDim.x >> 5) + (threadIdx.x >> 5);
  if (wid >= NN * 49) return;              // wave-uniform guard
  int lane = threadIdx.x & 31;
  int n = wid / 49, t = wid % 49;
  int tm = t / 7, tn = t % 7;
  const _Float16* Ap = P  + ((size_t)n * MQP + tm * 16 + (lane & 15)) * MSP;
  const _Float16* Bp = Qm + ((size_t)n * MQP + tn * 16 + (lane & 15)) * MSP;
  v8f acc = {};
  for (int k = 0; k < MSP; k += 32) {
    v16h av = load_fragA(Ap, k, lane);
    v16h bv = load_fragB(Bp, k, lane);
    acc = __builtin_amdgcn_wmma_f32_16x16x32_f16(false, av, false, bv,
                                                 (short)0, acc, false, false);
  }
  int r0  = tm * 16 + ((lane < 16) ? 0 : 8);
  int col = tn * 16 + (lane & 15);
  float* Wp = W + ((size_t)n * MQP + r0) * MQP + col;
#pragma unroll
  for (int i = 0; i < 8; ++i) Wp[(size_t)i * MQP] = acc[i];
}

// ---------------- K7: (I - 0.25 W) x = 1 + 0.5*rowsum(P); Neumann fixed point ----------------
// rho(0.25 W) = 0.25 (W column-stochastic) -> 32 iterations converge far below fp32 eps
__global__ void k_katz_solve(const _Float16* __restrict__ P, const float* __restrict__ W,
                             float* __restrict__ out) {
  int n = blockIdx.x;
  int tid = threadIdx.x;                  // 128 threads
  __shared__ float x[128];
  __shared__ float red[128];
  float rhs = 0.f;
  if (tid < MM) {
    const _Float16* Pp = P + ((size_t)n * MQP + tid) * MSP;
    float rs = 0.f;
    for (int s = 0; s < MS; ++s) rs += (float)Pp[s];
    rhs = 1.f + 0.5f * rs;
  }
  x[tid] = (tid < MM) ? rhs : 0.f;
  __syncthreads();
  const float* Wp = W + ((size_t)n * MQP + tid) * MQP;   // only deref'd when tid<100
  for (int it = 0; it < 32; ++it) {
    float acc = 0.f;
    if (tid < MM)
      for (int q2 = 0; q2 < MM; ++q2) acc += Wp[q2] * x[q2];
    float nx = rhs + 0.25f * acc;
    __syncthreads();
    if (tid < MM) x[tid] = nx;
    __syncthreads();
  }
  float katz = (tid < MM) ? (x[tid] - 1.f) : 0.f;
  red[tid] = katz;
  __syncthreads();
  for (int off = 64; off; off >>= 1) {
    if (tid < off) red[tid] += red[tid + off];
    __syncthreads();
  }
  float tot = red[0];
  if (tid < MM) out[(size_t)n * MM + tid] = katz / tot;
}

// ---------------- launch ----------------
extern "C" void kernel_launch(void* const* d_in, const int* in_sizes, int n_in,
                              void* d_out, int out_size, void* d_ws, size_t ws_size,
                              hipStream_t stream) {
  const float* sup = (const float*)d_in[0];   // (4,25,640,10,10)
  const float* qry = (const float*)d_in[1];   // (4,75,640,10,10)
  float* out = (float*)d_out;                 // (4,75,1,10,10) -> 30000 f32

  char* ws = (char*)d_ws;
  size_t off = 0;
  auto alloc = [&](size_t bytes) { size_t o = off; off += (bytes + 255) & ~(size_t)255; return o; };

  _Float16* qry_h = (_Float16*)(ws + alloc((size_t)NN * MQP * CH * 2));   // 49.2 MB
  _Float16* sup_h = (_Float16*)(ws + alloc((size_t)BB * MSP * CH * 2));   //  2.6 MB
  float*    supm  = (float*)   (ws + alloc((size_t)BB * NWAY * CH * MM * 4)); // 5.1 MB
  float*    S     = (float*)   (ws + alloc((size_t)NN * MQP * MSP * 4));  // 78.6 MB
  _Float16* Pm    = (_Float16*)(ws + alloc((size_t)NN * MQP * MSP * 2));  // 39.3 MB
  _Float16* Qm    = (_Float16*)(ws + alloc((size_t)NN * MQP * MSP * 2));  // 39.3 MB
  float*    W     = (float*)   (ws + alloc((size_t)NN * MQP * MQP * 4));  // 19.7 MB
  float*    rowm  = (float*)   (ws + alloc((size_t)NN * MQP * 4));
  float*    rowz  = (float*)   (ws + alloc((size_t)NN * MQP * 4));
  float*    colm  = (float*)   (ws + alloc((size_t)NN * MSP * 4));
  float*    colz  = (float*)   (ws + alloc((size_t)NN * MSP * 4));
  (void)ws_size; (void)in_sizes; (void)n_in; (void)out_size;

  k_prep_sup<<<BB * NWAY, 128, 0, stream>>>(sup, supm, sup_h);
  k_prep_qry<<<NN, 128, 0, stream>>>(qry, qry_h);

  // S GEMM: 300 * 4 * 8 = 9600 waves (each does a 32x64 strip), 8 waves / 256-thread block
  k_gemm_S<<<(NN * 32) / 8, 256, 0, stream>>>(qry_h, sup_h, S);

  k_row_stats<<<(NN * MM) / 8, 256, 0, stream>>>(S, rowm, rowz);
  k_col_stats<<<(NN * MSP) / 256, 256, 0, stream>>>(S, colm, colz);

  k_make_pq<<<(int)(((size_t)NN * MQP * MSP) / 256), 256, 0, stream>>>(
      S, rowm, rowz, colm, colz, Pm, Qm);

  // W GEMM: 300 * 49 = 14700 waves
  k_gemm_W<<<(NN * 49 + 7) / 8, 256, 0, stream>>>(Pm, Qm, W);

  k_katz_solve<<<NN, 128, 0, stream>>>(Pm, W, out);
}
